// LSTMnoise_80874234184287
// MI455X (gfx1250) — matrix operand order repaired and checked by reading, not compiled, not run on hardware
//
#include <hip/hip_runtime.h>
#include <math.h>

// ---------------- problem constants ----------------
#define H      512
#define INP    64
#define NL     4
#define LSEQ   4
#define TT     32
#define BB     512

// act buffer columns: [ x(64) | ht(512) | nhx0..nhx3 (4*512) ]  -> 2624
#define ACT_LD   2624
// ncy buffer columns: [ ct(512) | ncx0..ncx3 ] -> 2560
#define NCY_LD   2560

// d_out layout (floats): preds (29*512*64) | upreds (29*512) | sh (512*512) | sc (512*512)
#define OUT_PREDS   0
#define OUT_UPREDS  950272
#define OUT_SH      965120
#define OUT_SC      1227264

// GEMM tiling
#define BM    128          // block M tile
#define BN    64           // block N tile
#define LDSP  40           // padded LDS row stride (halves); rows are 32 halves

// async-copy path (CDNA5 GLOBAL_LOAD_ASYNC_TO_LDS), feature-gated
#if defined(__has_builtin)
#  if __has_builtin(__builtin_amdgcn_global_load_async_to_lds_b128) && \
      __has_builtin(__builtin_amdgcn_s_wait_asynccnt)
#    define USE_ASYNC_LDS 1
#  endif
#endif
#ifndef USE_ASYNC_LDS
#  define USE_ASYNC_LDS 0
#endif

typedef __attribute__((ext_vector_type(16))) __bf16 v16bf;
typedef __attribute__((ext_vector_type(8)))  __bf16 v8bf;
typedef __attribute__((ext_vector_type(8)))  float  v8f;

#if USE_ASYNC_LDS
// builtin signature (from hipcc diagnostic): params are pointers to
// 'int __attribute__((vector_size(16)))' in AS1 (global) / AS3 (LDS).
typedef __attribute__((__vector_size__(4 * sizeof(int)))) int v4i_;
typedef __attribute__((address_space(1))) v4i_* gptr_v4i;
typedef __attribute__((address_space(3))) v4i_* lptr_v4i;
#endif

union FragBF { v16bf v; v8bf h[2]; };

__device__ __forceinline__ float sigm_(float x) { return 1.0f / (1.0f + __expf(-x)); }

__device__ __forceinline__ v8f wmma_bf16_(const FragBF& a, const FragBF& b, v8f c) {
    return __builtin_amdgcn_wmma_f32_16x16x32_bf16(false, a.v, false, b.v,
                                                   (short)0, c, false, false);
}

// ---------------------------------------------------------------------------
// Tiled bf16 WMMA GEMM:  C[m,n] (+)= sum_k A[m,k]*W[n,k] (+ bias[n]) (, relu)
// Block = 256 threads = 8 waves. Block tile 128(M) x 64(N); wave tile 32x32
// (4 WMMAs / K-step / wave). K stepped by 32. Double-buffered LDS staging:
// async copy (GLOBAL_LOAD_ASYNC_TO_LDS_B128 + s_wait_asynccnt) when the
// toolchain exposes it, else global_load_b128 + ds_store_b128. One workgroup
// barrier per K-step.
// flags: bit0 = accumulate into existing C, bit1 = relu.
// ---------------------------------------------------------------------------
__global__ __launch_bounds__(256)
void gemm_bf16_wmma(const __bf16* __restrict__ A, int lda,
                    const __bf16* __restrict__ W, int ldw,
                    const float* __restrict__ bias,
                    float* __restrict__ C, int ldc,
                    __bf16* __restrict__ Obf, int ldo,
                    int K, int flags)
{
    __shared__ __align__(16) __bf16 sA[2 * BM * LDSP];   // 2 x 128 x 32 (padded)
    __shared__ __align__(16) __bf16 sW[2 * BN * LDSP];   // 2 x  64 x 32 (padded)

    const int tid   = threadIdx.x;
    const int mBase = blockIdx.y * BM;
    const int nBase = blockIdx.x * BN;
    const int w     = tid >> 5;       // wave id 0..7
    const int lane  = tid & 31;
    const int lh    = lane >> 4;      // lane-half (0/1)
    const int l16   = lane & 15;
    const int wm    = w & 3;          // 4 M sub-tiles of 32
    const int wn    = w >> 2;         // 2 N sub-tiles of 32

    v8f acc00 = {}, acc01 = {}, acc10 = {}, acc11 = {};

    // staging: (BM+BN)=192 rows x 4 chunks(16B) = 768 chunks; 3 per thread
    auto stage = [&](int kb, int db) {
#pragma unroll
        for (int j = 0; j < 3; ++j) {
            const int c   = tid + j * 256;
            const int row = c >> 2;
            const int col = (c & 3) << 3;
            const __bf16* gsrc;
            __bf16* ldst;
            if (row < BM) {
                gsrc = &A[(size_t)(mBase + row) * lda + kb + col];
                ldst = &sA[(db * BM + row) * LDSP + col];
            } else {
                const int r = row - BM;
                gsrc = &W[(size_t)(nBase + r) * ldw + kb + col];
                ldst = &sW[(db * BN + r) * LDSP + col];
            }
#if USE_ASYNC_LDS
            __builtin_amdgcn_global_load_async_to_lds_b128(
                (gptr_v4i)(v4i_*)(void*)(const void*)gsrc,
                (lptr_v4i)(v4i_*)(void*)ldst, 0, 0);
#else
            *(uint4*)ldst = *(const uint4*)gsrc;
#endif
        }
    };

    auto compute = [&](int db) {
        const __bf16* aB = &sA[db * BM * LDSP];
        const __bf16* wB = &sW[db * BN * LDSP];
        FragBF a0, a1, b0, b1;
        // A fragments: lane(l16)=row, halves 0-7 -> K lh*8.., 8-15 -> K 16+lh*8..
        const int r0 = (wm * 32 + l16) * LDSP + lh * 8;
        a0.h[0] = *(const v8bf*)(&aB[r0]);
        a0.h[1] = *(const v8bf*)(&aB[r0 + 16]);
        const int r1 = (wm * 32 + 16 + l16) * LDSP + lh * 8;
        a1.h[0] = *(const v8bf*)(&aB[r1]);
        a1.h[1] = *(const v8bf*)(&aB[r1 + 16]);
        // B fragments: lane(l16)=col n, halves 0-7 -> K lh*8.., 8-15 -> K 16+lh*8..
        const int c0 = (wn * 32 + l16) * LDSP + lh * 8;
        b0.h[0] = *(const v8bf*)(&wB[c0]);
        b0.h[1] = *(const v8bf*)(&wB[c0 + 16]);
        const int c1 = (wn * 32 + 16 + l16) * LDSP + lh * 8;
        b1.h[0] = *(const v8bf*)(&wB[c1]);
        b1.h[1] = *(const v8bf*)(&wB[c1 + 16]);
        acc00 = wmma_bf16_(a0, b0, acc00);
        acc01 = wmma_bf16_(a0, b1, acc01);
        acc10 = wmma_bf16_(a1, b0, acc10);
        acc11 = wmma_bf16_(a1, b1, acc11);
    };

    stage(0, 0);
#if USE_ASYNC_LDS
    __builtin_amdgcn_s_wait_asynccnt(0);
#endif
    __syncthreads();

    int db = 0;
    for (int kb = 0; kb < K; kb += 32) {
        if (kb + 32 < K) {
            stage(kb + 32, db ^ 1);
            __builtin_prefetch(&W[(size_t)(nBase + (tid >> 2)) * ldw + kb + 64], 0, 1);
        }
        compute(db);
#if USE_ASYNC_LDS
        __builtin_amdgcn_s_wait_asynccnt(0);
#endif
        __syncthreads();
        db ^= 1;
    }

    // epilogue: C/D layout -> vgpr r holds m = r + 8*lh, n = l16 (+16)
    const int n0 = nBase + wn * 32 + l16;
#pragma unroll
    for (int s = 0; s < 2; ++s) {
        const v8f& aL = s ? acc10 : acc00;
        const v8f& aR = s ? acc11 : acc01;
        const int mrow = mBase + wm * 32 + s * 16 + (lh << 3);
#pragma unroll
        for (int r = 0; r < 8; ++r) {
            float v0 = aL[r];
            float v1 = aR[r];
            if (bias) { v0 += bias[n0]; v1 += bias[n0 + 16]; }
            if (flags & 1) {
                v0 += C[(size_t)(mrow + r) * ldc + n0];
                v1 += C[(size_t)(mrow + r) * ldc + n0 + 16];
            }
            if (flags & 2) { v0 = fmaxf(v0, 0.0f); v1 = fmaxf(v1, 0.0f); }
            if (C) {
                C[(size_t)(mrow + r) * ldc + n0]      = v0;
                C[(size_t)(mrow + r) * ldc + n0 + 16] = v1;
            }
            if (Obf) {
                Obf[(size_t)(mrow + r) * ldo + n0]      = (__bf16)v0;
                Obf[(size_t)(mrow + r) * ldo + n0 + 16] = (__bf16)v1;
            }
        }
    }
}

// ---------------- weight prep ----------------
__global__ void k_conv_bf16(__bf16* __restrict__ dst, const float* __restrict__ src, int n) {
    int g = blockIdx.x * 256 + threadIdx.x;
    if (g < n) dst[g] = (__bf16)src[g];
}

// combined [Wih | Whh] per layer, row-major 2048 x Kc
__global__ void k_prep_wcomb(__bf16* __restrict__ dst, const float* __restrict__ Wih,
                             const float* __restrict__ Whh, int Kc, int Kh, int n) {
    int g = blockIdx.x * 256 + threadIdx.x;
    if (g >= n) return;
    int o = g / Kc;
    int k = g - o * Kc;
    float v = (k < INP) ? Wih[o * INP + k] : Whh[(size_t)o * Kh + (k - INP)];
    dst[g] = (__bf16)v;
}

__global__ void k_bcomb(float* __restrict__ dst, const float* __restrict__ b0,
                        const float* __restrict__ b1, int n) {
    int g = blockIdx.x * 256 + threadIdx.x;
    if (g < n) dst[g] = b0[g] + b1[g];
}

// ---------------- state init (ht0 = ct0 = stage = 0) ----------------
__global__ void k_init(__bf16* __restrict__ act, __bf16* __restrict__ ncy,
                       float* __restrict__ cprev, __bf16* __restrict__ stage) {
    int g = blockIdx.x * 256 + threadIdx.x;   // 512*512 threads
    int b = g >> 9, h = g & 511;
    act[(size_t)b * ACT_LD + INP + h] = (__bf16)0.0f;
    ncy[(size_t)b * NCY_LD + h]       = (__bf16)0.0f;
    cprev[g] = 0.0f;
    for (int s = 0; s < LSEQ; ++s) stage[(size_t)s * (BB * H) + g] = (__bf16)0.0f;
}

// ---------------- per-timestep elementwise kernels ----------------
__global__ void k_prep_x(const float* __restrict__ inputs, const float* __restrict__ noise,
                         const int* __restrict__ idx, __bf16* __restrict__ act,
                         float* __restrict__ pred, int t) {
    int b = blockIdx.x, j = threadIdx.x;  // 512 x 64
    float v = inputs[((size_t)t * BB + b) * INP + j];
    if (j == 0) v -= noise[t + idx[b]];
    pred[b * INP + j] = v;                       // pred starts as istate
    act[(size_t)b * ACT_LD + j] = (__bf16)v;     // x columns
}

__global__ void k_cell_ew(const float* __restrict__ gates, const float* __restrict__ peep,
                          float* __restrict__ cprev, __bf16* __restrict__ act,
                          __bf16* __restrict__ ncy, int layer) {
    int g = blockIdx.x * 256 + threadIdx.x;   // 512*512
    int b = g >> 9, h = g & 511;
    const float* gr = gates + (size_t)b * (4 * H);
    const float* pr = peep + (size_t)b * (3 * H);
    float i_g = sigm_(gr[h] + pr[h]);
    float f_g = sigm_(gr[H + h] + pr[H + h]);
    float c_g = tanhf(gr[2 * H + h]);
    float o_g = sigm_(gr[3 * H + h] + pr[2 * H + h]);
    float cx  = cprev[g];
    float ncx = f_g * cx + i_g * c_g;
    float nhx = o_g * tanhf(ncx);
    cprev[g] = ncx;
    act[(size_t)b * ACT_LD + INP + (layer + 1) * H + h] = (__bf16)nhx;
    ncy[(size_t)b * NCY_LD + (layer + 1) * H + h]       = (__bf16)ncx;
}

__global__ void k_advance(const float* __restrict__ hout, const float* __restrict__ cprev,
                          __bf16* __restrict__ act, __bf16* __restrict__ ncy,
                          __bf16* __restrict__ stage, float* __restrict__ out,
                          const int* __restrict__ npred, int t) {
    int g = blockIdx.x * 256 + threadIdx.x;   // 512*512
    int b = g >> 9, h = g & 511;
    float hv = hout[g];
    act[(size_t)b * ACT_LD + INP + h] = (__bf16)hv;          // next step's ht columns
    ncy[(size_t)b * NCY_LD + h]       = (__bf16)cprev[g];    // next step's ct columns
    stage[(size_t)(t & 3) * (BB * H) + g] = (__bf16)hv;      // stage ring slot
    if (t == npred[0] - 1) {
        out[OUT_SH + g] = hv;
        out[OUT_SC + g] = cprev[g];
    }
}

__global__ void k_pred_final(const float* __restrict__ pred, const float* __restrict__ noise,
                             const int* __restrict__ idx, const int* __restrict__ npred,
                             float* __restrict__ out, int t) {
    int b = blockIdx.x, j = threadIdx.x;  // 512 x 64
    float v = pred[b * INP + j];
    if (j == 0) v += noise[npred[0] + t + idx[b]];
    if (t >= LSEQ - 1) {
        out[OUT_PREDS + ((size_t)(t - (LSEQ - 1)) * BB + b) * INP + j] = v;
        if (j == 0) out[OUT_UPREDS + (size_t)(t - (LSEQ - 1)) * BB + b] = v;
    }
}

// ---------------------------------------------------------------------------
extern "C" void kernel_launch(void* const* d_in, const int* in_sizes, int n_in,
                              void* d_out, int out_size, void* d_ws, size_t ws_size,
                              hipStream_t stream) {
    (void)in_sizes; (void)n_in; (void)out_size; (void)ws_size;

    // ---- input pointers (setup_inputs dict order, depth-first) ----
    const float* inputs = (const float*)d_in[0];
    const float* noise  = (const float*)d_in[1];
    const float* Wih[NL], *bih[NL], *Whh[NL], *bhh[NL], *Wch[NL], *bch[NL], *Whl[NL], *bhl[NL];
    for (int i = 0; i < NL; ++i) {
        Wih[i] = (const float*)d_in[2 + i];
        bih[i] = (const float*)d_in[6 + i];
        Whh[i] = (const float*)d_in[10 + i];
        bhh[i] = (const float*)d_in[14 + i];
        Wch[i] = (const float*)d_in[18 + i];
        bch[i] = (const float*)d_in[22 + i];
        Whl[i] = (const float*)d_in[26 + i];
        bhl[i] = (const float*)d_in[30 + i];
    }
    const float* Wl1 = (const float*)d_in[34];   // (4, 5120, 512)
    const float* bl1 = (const float*)d_in[35];   // (4, 5120)
    const float* Wl2 = (const float*)d_in[36];   // (4, 64, 5120)
    const float* bl2 = (const float*)d_in[37];   // (4, 64)
    const int*   idx   = (const int*)d_in[38];
    const int*   npred = (const int*)d_in[39];
    float* out = (float*)d_out;

    // ---- workspace carve ----
    size_t off = 0;
    char* base = (char*)d_ws;
    auto carve = [&](size_t bytes) -> char* {
        off = (off + 255) & ~(size_t)255;
        char* p = base + off;
        off += bytes;
        return p;
    };
    const int Kc[NL] = {INP + 1 * H, INP + 2 * H, INP + 3 * H, INP + 4 * H}; // 576,1088,1600,2112

    __bf16* wcomb[NL]; float* bcomb[NL]; __bf16* wchb[NL]; __bf16* whlb[NL];
    for (int i = 0; i < NL; ++i) wcomb[i] = (__bf16*)carve((size_t)4 * H * Kc[i] * 2);
    for (int i = 0; i < NL; ++i) bcomb[i] = (float*)carve((size_t)4 * H * 4);
    for (int i = 0; i < NL; ++i) wchb[i]  = (__bf16*)carve((size_t)3 * H * (i + 1) * H * 2);
    for (int i = 0; i < NL; ++i) whlb[i]  = (__bf16*)carve((size_t)H * H * 2);
    __bf16* wl1b = (__bf16*)carve((size_t)LSEQ * 10 * H * H * 2);
    __bf16* wl2b = (__bf16*)carve((size_t)LSEQ * INP * 10 * H * 2);

    __bf16* act   = (__bf16*)carve((size_t)BB * ACT_LD * 2);
    __bf16* ncy   = (__bf16*)carve((size_t)BB * NCY_LD * 2);
    __bf16* stage = (__bf16*)carve((size_t)LSEQ * BB * H * 2);
    __bf16* h1    = (__bf16*)carve((size_t)BB * 10 * H * 2);
    float*  gates = (float*)carve((size_t)BB * 4 * H * 4);
    float*  peep  = (float*)carve((size_t)BB * 3 * H * 4);
    float*  hout  = (float*)carve((size_t)BB * H * 4);
    float*  cprev = (float*)carve((size_t)BB * H * 4);
    float*  pred  = (float*)carve((size_t)BB * INP * 4);

    // ---- weight conversion (bf16, combined ih|hh per layer) ----
    for (int i = 0; i < NL; ++i) {
        int n = 4 * H * Kc[i];
        k_prep_wcomb<<<(n + 255) / 256, 256, 0, stream>>>(wcomb[i], Wih[i], Whh[i],
                                                          Kc[i], (i + 1) * H, n);
        k_bcomb<<<(4 * H + 255) / 256, 256, 0, stream>>>(bcomb[i], bih[i], bhh[i], 4 * H);
        int nch = 3 * H * (i + 1) * H;
        k_conv_bf16<<<(nch + 255) / 256, 256, 0, stream>>>(wchb[i], Wch[i], nch);
        k_conv_bf16<<<(H * H + 255) / 256, 256, 0, stream>>>(whlb[i], Whl[i], H * H);
    }
    {
        int n1 = LSEQ * 10 * H * H;
        k_conv_bf16<<<(n1 + 255) / 256, 256, 0, stream>>>(wl1b, Wl1, n1);
        int n2 = LSEQ * INP * 10 * H;
        k_conv_bf16<<<(n2 + 255) / 256, 256, 0, stream>>>(wl2b, Wl2, n2);
    }
    // zero initial state
    k_init<<<(BB * H) / 256, 256, 0, stream>>>(act, ncy, cprev, stage);

    // ---- time loop ----
    for (int t = 0; t < TT; ++t) {
        k_prep_x<<<BB, INP, 0, stream>>>(inputs, noise, idx, act, pred, t);

        for (int i = 0; i < NL; ++i) {
            // gates = [x|nhy] @ Wcomb_i^T + (bih+bhh)
            gemm_bf16_wmma<<<dim3((4 * H) / BN, BB / BM), 256, 0, stream>>>(
                act, ACT_LD, wcomb[i], Kc[i], bcomb[i], gates, 4 * H,
                (__bf16*)nullptr, 0, Kc[i], 0);
            // peep = ncy @ Wch_i^T + bch
            gemm_bf16_wmma<<<dim3((3 * H) / BN, BB / BM), 256, 0, stream>>>(
                ncy, NCY_LD, wchb[i], (i + 1) * H, bch[i], peep, 3 * H,
                (__bf16*)nullptr, 0, (i + 1) * H, 0);
            // elementwise cell update; appends nhx/ncx columns
            k_cell_ew<<<(BB * H) / 256, 256, 0, stream>>>(gates, peep, cprev, act, ncy, i);
            // hout (+)= nhx @ Whl_i^T + bhl
            gemm_bf16_wmma<<<dim3(H / BN, BB / BM), 256, 0, stream>>>(
                act + INP + (i + 1) * H, ACT_LD, whlb[i], H, bhl[i], hout, H,
                (__bf16*)nullptr, 0, H, (i > 0) ? 1 : 0);
        }

        k_advance<<<(BB * H) / 256, 256, 0, stream>>>(hout, cprev, act, ncy, stage,
                                                      out, npred, t);

        // heads: stage[l] = ht(t-3+l) lives in ring slot (t+1+l)&3
        for (int l = 0; l < LSEQ; ++l) {
            int slot = (t + 1 + l) & 3;
            // h1 = relu(stage_l @ Wl1_l^T + bl1_l)   (bf16 out only)
            gemm_bf16_wmma<<<dim3((10 * H) / BN, BB / BM), 256, 0, stream>>>(
                stage + (size_t)slot * BB * H, H, wl1b + (size_t)l * 10 * H * H, H,
                bl1 + l * 10 * H, (float*)nullptr, 0, h1, 10 * H, H, 2);
            // pred += h1 @ Wl2_l^T + bl2_l
            gemm_bf16_wmma<<<dim3(INP / BN, BB / BM), 256, 0, stream>>>(
                h1, 10 * H, wl2b + (size_t)l * INP * 10 * H, 10 * H,
                bl2 + l * INP, pred, INP, (__bf16*)nullptr, 0, 10 * H, 1);
        }

        k_pred_final<<<BB, INP, 0, stream>>>(pred, noise, idx, npred, out, t);
    }
}
